// SSIA_15908558864697
// MI455X (gfx1250) — matrix operand order repaired and checked by reading, not compiled, not run on hardware
//
#include <hip/hip_runtime.h>

typedef __attribute__((ext_vector_type(16))) _Float16 v16h;
typedef __attribute__((ext_vector_type(8)))  float    v8f;

constexpr int B_  = 2;
constexpr int C_  = 128;
constexpr int HW_ = 4096;
constexpr int Hh_ = 64;
constexpr int Wh_ = 64;
constexpr int HEADS_ = 4;
constexpr int CH_ = 32;       // channels per head
constexpr int QC_ = 5 * C_;   // 640 qkv channels

// ---------------------------------------------------------------------------
// BiasFree LayerNorm over channel dim, fused f16 transpose:
// in  x  fp32 [b][C][HW]  ->  out yt f16 [b][HW][C]  (pixel-major for GEMM B)
// ---------------------------------------------------------------------------
__global__ void ln_ft_k(const float* __restrict__ x, const float* __restrict__ w,
                        _Float16* __restrict__ yt) {
  int p = blockIdx.x * blockDim.x + threadIdx.x;
  if (p >= B_ * HW_) return;
  int b = p / HW_, n = p % HW_;
  const float* xp = x + (size_t)b * C_ * HW_ + n;
  float s = 0.f, s2 = 0.f;
  for (int c = 0; c < C_; ++c) { float v = xp[(size_t)c * HW_]; s += v; s2 += v * v; }
  float mean = s * (1.0f / C_);
  float var  = s2 * (1.0f / C_) - mean * mean;
  float inv  = rsqrtf(var + 1e-5f);
  _Float16* yp = yt + (size_t)p * C_;
  for (int c = 0; c < C_; ++c) yp[c] = (_Float16)(xp[(size_t)c * HW_] * inv * w[c]);
}

// ---------------------------------------------------------------------------
// f32 -> f16 elementwise (weights)
// ---------------------------------------------------------------------------
__global__ void cvt_k(const float* __restrict__ in, _Float16* __restrict__ out, int n) {
  int i = blockIdx.x * blockDim.x + threadIdx.x;
  if (i < n) out[i] = (_Float16)in[i];
}

// ---------------------------------------------------------------------------
// WMMA f16 GEMM, K fixed at 128 (fully unrolled -> 16 WMMA):
//   C[z][M][N] = A[M][128] * Bt[z][N][128]^T (+bias[M])
// A row-major, Bt N-major (so every fragment load is contiguous per lane).
// One wave/block computes a 16(M) x 64(N) tile: 4 accumulators share 1 A frag.
// A frag: lane ln = row, halves K {16hi..16hi+15}.
// B frag: lane ln = col, halves K {16hi..16hi+15} (contiguous in Bt row).
// C frag: lane col = ln, VGPR r -> row r + 8hi.
// ---------------------------------------------------------------------------
__global__ void gemm_k(const _Float16* __restrict__ A, const _Float16* __restrict__ Bt,
                       float* __restrict__ Cm, const float* __restrict__ bias,
                       int M, int N) {
  constexpr int KK = 128;
  int nt = blockIdx.x * 64;
  int mt = blockIdx.y * 16;
  size_t boff = (size_t)blockIdx.z * N * KK;
  size_t coff = (size_t)blockIdx.z * M * N;
  int lane = threadIdx.x & 31;
  int hi = lane >> 4, ln = lane & 15;

  v8f acc0 = {}, acc1 = {}, acc2 = {}, acc3 = {};
  const _Float16* ap = A + (size_t)(mt + ln) * KK;
  const _Float16* bp0 = Bt + boff + (size_t)(nt + ln) * KK + hi * 16;
  const _Float16* bp1 = bp0 + (size_t)16 * KK;
  const _Float16* bp2 = bp0 + (size_t)32 * KK;
  const _Float16* bp3 = bp0 + (size_t)48 * KK;

#pragma unroll
  for (int k0 = 0; k0 < KK; k0 += 32) {
    v16h a;
    int kb = hi * 16;
#pragma unroll
    for (int i = 0; i < 8; ++i) { a[i] = ap[k0 + kb + i]; a[8 + i] = ap[k0 + kb + 8 + i]; }
    v16h b0, b1, b2, b3;
#pragma unroll
    for (int i = 0; i < 16; ++i) {
      b0[i] = bp0[k0 + i];
      b1[i] = bp1[k0 + i];
      b2[i] = bp2[k0 + i];
      b3[i] = bp3[k0 + i];
    }
    acc0 = __builtin_amdgcn_wmma_f32_16x16x32_f16(false, a, false, b0, (short)0, acc0, false, false);
    acc1 = __builtin_amdgcn_wmma_f32_16x16x32_f16(false, a, false, b1, (short)0, acc1, false, false);
    acc2 = __builtin_amdgcn_wmma_f32_16x16x32_f16(false, a, false, b2, (short)0, acc2, false, false);
    acc3 = __builtin_amdgcn_wmma_f32_16x16x32_f16(false, a, false, b3, (short)0, acc3, false, false);
  }

  float bv[8];
  if (bias) {
#pragma unroll
    for (int r = 0; r < 8; ++r) bv[r] = bias[mt + r + 8 * hi];
  } else {
#pragma unroll
    for (int r = 0; r < 8; ++r) bv[r] = 0.f;
  }
#pragma unroll
  for (int r = 0; r < 8; ++r) {
    size_t row = coff + (size_t)(mt + r + 8 * hi) * N + nt + ln;
    Cm[row]      = acc0[r] + bv[r];
    Cm[row + 16] = acc1[r] + bv[r];
    Cm[row + 32] = acc2[r] + bv[r];
    Cm[row + 48] = acc3[r] + bv[r];
  }
}

// ---------------------------------------------------------------------------
// Depthwise 3x3, SAME padding, cross-correlation (XLA conv semantics)
// ---------------------------------------------------------------------------
__global__ void dwconv_k(const float* __restrict__ in, const float* __restrict__ wdw,
                         float* __restrict__ out) {
  long i = (long)blockIdx.x * blockDim.x + threadIdx.x;
  if (i >= (long)B_ * QC_ * HW_) return;
  int n = (int)(i % HW_);
  int ch = (int)((i / HW_) % QC_);
  int y = n >> 6, x = n & 63;
  const float* wp = wdw + (size_t)ch * 9;
  const float* ip = in + (i - n);
  float acc = 0.f;
  for (int dy = -1; dy <= 1; ++dy) {
    int yy = y + dy;
    if (yy < 0 || yy >= Hh_) continue;
    for (int dx = -1; dx <= 1; ++dx) {
      int xx = x + dx;
      if (xx < 0 || xx >= Wh_) continue;
      acc += ip[yy * Wh_ + xx] * wp[(dy + 1) * 3 + (dx + 1)];
    }
  }
  out[i] = acc;
}

// ---------------------------------------------------------------------------
// Per-channel inverse L2 norm over tokens for q1,k1,q2,k2.
// norms[b][t][c], t in {0:q1, 1:k1, 2:q2, 3:k2}
// ---------------------------------------------------------------------------
__global__ void norms_k(const float* __restrict__ dwc, float* __restrict__ norms) {
  int idx = blockIdx.x;
  int c = idx % C_;
  int t = (idx / C_) % 4;
  int b = idx / (4 * C_);
  const int ofs[4] = {0, C_, 3 * C_, 4 * C_};
  const float* p = dwc + ((size_t)b * QC_ + ofs[t] + c) * HW_;
  __shared__ float red[256];
  int tid = threadIdx.x;
  float s = 0.f;
  for (int n = tid; n < HW_; n += 256) { float v = p[n]; s += v * v; }
  red[tid] = s;
  __syncthreads();
  for (int st = 128; st > 0; st >>= 1) {
    if (tid < st) red[tid] += red[tid + st];
    __syncthreads();
  }
  if (tid == 0) norms[idx] = 1.0f / fmaxf(sqrtf(red[0]), 1e-12f);
}

// ---------------------------------------------------------------------------
// Pack q1 (x inv_norm x temperature), k1 (x inv_norm), v into f16 [bh][n][32]
// ---------------------------------------------------------------------------
__global__ void pack_k(const float* __restrict__ dwc, const float* __restrict__ norms,
                       const float* __restrict__ temp,
                       _Float16* __restrict__ q1h, _Float16* __restrict__ k1h,
                       _Float16* __restrict__ vh) {
  long i = (long)blockIdx.x * blockDim.x + threadIdx.x;
  if (i >= (long)B_ * HEADS_ * HW_ * CH_) return;
  int c  = (int)(i % CH_);
  int n  = (int)((i / CH_) % HW_);
  int hd = (int)((i / ((long)CH_ * HW_)) % HEADS_);
  int b  = (int)(i / ((long)CH_ * HW_ * HEADS_));
  int cg = hd * CH_ + c;
  size_t base = (size_t)b * QC_ * HW_;
  float q1 = dwc[base + (size_t)(0 * C_ + cg) * HW_ + n];
  float k1 = dwc[base + (size_t)(1 * C_ + cg) * HW_ + n];
  float v  = dwc[base + (size_t)(2 * C_ + cg) * HW_ + n];
  float t  = temp[hd];
  float iq = norms[(b * 4 + 0) * C_ + cg];
  float ik = norms[(b * 4 + 1) * C_ + cg];
  q1h[i] = (_Float16)(q1 * iq * t);
  k1h[i] = (_Float16)(k1 * ik);
  vh[i]  = (_Float16)v;
}

// ---------------------------------------------------------------------------
// Channel attention: attn2[bh][c][d] = softmax_d( temp * q2n^T k2n ) (32x32)
// ---------------------------------------------------------------------------
__global__ void chan_attn_k(const float* __restrict__ dwc, const float* __restrict__ norms,
                            const float* __restrict__ temp, float* __restrict__ attn2) {
  int bh = blockIdx.x;
  int b = bh >> 2, hd = bh & 3;
  const float* q2p = dwc + ((size_t)b * QC_ + 3 * C_ + hd * CH_) * HW_;
  const float* k2p = dwc + ((size_t)b * QC_ + 4 * C_ + hd * CH_) * HW_;
  __shared__ float qs[CH_][64];
  __shared__ float ks[CH_][64];
  __shared__ float G[CH_][CH_ + 1];
  int tid = threadIdx.x;
  float acc[4] = {0.f, 0.f, 0.f, 0.f};
  for (int n0 = 0; n0 < HW_; n0 += 64) {
    for (int i = tid; i < CH_ * 64; i += 256) {
      int cc = i >> 6, nn = i & 63;
      qs[cc][nn] = q2p[(size_t)cc * HW_ + n0 + nn];
      ks[cc][nn] = k2p[(size_t)cc * HW_ + n0 + nn];
    }
    __syncthreads();
    for (int u = 0; u < 4; ++u) {
      int pidx = tid + u * 256;
      int c = pidx >> 5, d = pidx & 31;
      float a = acc[u];
      for (int nn = 0; nn < 64; ++nn) a += qs[c][nn] * ks[d][nn];
      acc[u] = a;
    }
    __syncthreads();
  }
  float tmp = temp[hd];
  for (int u = 0; u < 4; ++u) {
    int pidx = tid + u * 256;
    int c = pidx >> 5, d = pidx & 31;
    float iq = norms[(b * 4 + 2) * C_ + hd * CH_ + c];
    float ik = norms[(b * 4 + 3) * C_ + hd * CH_ + d];
    G[c][d] = acc[u] * tmp * iq * ik;
  }
  __syncthreads();
  if (tid < CH_) {
    int c = tid;
    float mx = -1e30f;
    for (int d = 0; d < CH_; ++d) mx = fmaxf(mx, G[c][d]);
    float s = 0.f;
    float e[CH_];
    for (int d = 0; d < CH_; ++d) { e[d] = __expf(G[c][d] - mx); s += e[d]; }
    float invs = 1.0f / s;
    for (int d = 0; d < CH_; ++d)
      attn2[((size_t)bh * CH_ + c) * CH_ + d] = e[d] * invs;
  }
}

// ---------------------------------------------------------------------------
// out2[bh][n][d] = sum_c v[n,c] * attn2[c,d]
// ---------------------------------------------------------------------------
__global__ void out2_k(const _Float16* __restrict__ vh, const float* __restrict__ attn2,
                       float* __restrict__ out2) {
  int bh = blockIdx.y;
  __shared__ float a2[CH_][CH_];
  int tid = threadIdx.x;
  for (int i = tid; i < CH_ * CH_; i += 256)
    a2[i >> 5][i & 31] = attn2[(size_t)bh * CH_ * CH_ + i];
  __syncthreads();
  int i = blockIdx.x * 256 + tid;
  int d = i & 31, n = i >> 5;
  const _Float16* vp = vh + ((size_t)bh * HW_ + n) * CH_;
  float acc = 0.f;
  for (int c = 0; c < CH_; ++c) acc += (float)vp[c] * a2[c][d];
  out2[(size_t)bh * HW_ * CH_ + i] = acc;
}

// ---------------------------------------------------------------------------
// Flash attention (spatial): 128 query rows/block (8 waves x 16), 32-key tiles
// staged in LDS (V transposed so all fragment reads are contiguous), online
// softmax via xor-shuffle row reductions, P re-laid through LDS.
// Adds channel-attn out2 and writes pixel-major f16 [b][n][C] for next GEMM.
// ---------------------------------------------------------------------------
__global__ void flash_k(const _Float16* __restrict__ q, const _Float16* __restrict__ k,
                        const _Float16* __restrict__ v, const float* __restrict__ out2,
                        _Float16* __restrict__ outt) {
  constexpr int NT = 32;        // keys per tile
  constexpr int VP = NT + 2;    // padded row stride for transposed V
  const int bh = blockIdx.y;
  const int b = bh >> 2, hd = bh & 3;
  const int q0 = blockIdx.x * 128;
  const int tid = threadIdx.x;
  const int wave = tid >> 5;
  const int lane = tid & 31;
  const int hi = lane >> 4;
  const int ln = lane & 15;

  __shared__ _Float16 kt[NT * CH_];      // [key][ch]
  __shared__ _Float16 vtt[CH_][VP];      // [ch][key] (transposed)
  __shared__ _Float16 pbuf[8][16 * NT];

  const _Float16* qb = q + (size_t)bh * HW_ * CH_;
  const _Float16* kb = k + (size_t)bh * HW_ * CH_;
  const _Float16* vb = v + (size_t)bh * HW_ * CH_;

  // Q A-fragment
  v16h qa;
  {
    const _Float16* qp = qb + (size_t)(q0 + wave * 16 + ln) * CH_;
    int kbse = hi * 16;
#pragma unroll
    for (int i = 0; i < 8; ++i) { qa[i] = qp[kbse + i]; qa[8 + i] = qp[kbse + 8 + i]; }
  }

  v8f o0 = {}, o1 = {};
  float mrow[8], lrow[8];
#pragma unroll
  for (int r = 0; r < 8; ++r) { mrow[r] = -1e30f; lrow[r] = 0.f; }

  const int NITER = HW_ / NT;
  for (int j = 0; j < NITER; ++j) {
    // cooperative tile load: K row-major (b128 copies), V transposed
    if (tid < 128) {
      ((uint4*)kt)[tid] = ((const uint4*)(kb + (size_t)j * NT * CH_))[tid];
    } else {
      int tt = tid - 128;                // 0..127 ; 4 chunks of 8 ch per key
      int key = tt >> 2, c8 = (tt & 3) * 8;
      union { uint4 u; _Float16 h[8]; } tmp;
      tmp.u = ((const uint4*)(vb + (size_t)j * NT * CH_))[tt];
#pragma unroll
      for (int i = 0; i < 8; ++i) vtt[c8 + i][key] = tmp.h[i];
    }
    if (j + 1 < NITER) {
      __builtin_prefetch(kb + (size_t)(j + 1) * NT * CH_, 0, 0);
      __builtin_prefetch(vb + (size_t)(j + 1) * NT * CH_, 0, 0);
    }
    __syncthreads();

    // K B-fragments: lane col = key, contiguous ch halves
    v16h b0, b1;
    {
      int khalf = hi * 16;
      const _Float16* p0 = &kt[ln * CH_ + khalf];
      const _Float16* p1 = &kt[(16 + ln) * CH_ + khalf];
#pragma unroll
      for (int i = 0; i < 16; ++i) { b0[i] = p0[i]; b1[i] = p1[i]; }
    }
    v8f zc = {};
    v8f s0 = __builtin_amdgcn_wmma_f32_16x16x32_f16(false, qa, false, b0, (short)0, zc, false, false);
    v8f s1 = __builtin_amdgcn_wmma_f32_16x16x32_f16(false, qa, false, b1, (short)0, zc, false, false);

    // online softmax: row stats via xor reduce across the 16 lanes per half
#pragma unroll
    for (int r = 0; r < 8; ++r) {
      float mx = fmaxf(s0[r], s1[r]);
      for (int d = 1; d < 16; d <<= 1) mx = fmaxf(mx, __shfl_xor(mx, d, 32));
      float mnew = fmaxf(mrow[r], mx);
      float scale = __expf(mrow[r] - mnew);
      float p0 = __expf(s0[r] - mnew);
      float p1 = __expf(s1[r] - mnew);
      float rs = p0 + p1;
      for (int d = 1; d < 16; d <<= 1) rs += __shfl_xor(rs, d, 32);
      lrow[r] = lrow[r] * scale + rs;
      mrow[r] = mnew;
      o0[r] *= scale;
      o1[r] *= scale;
      s0[r] = p0;
      s1[r] = p1;
    }

    // P: C layout -> A layout through LDS (wave-local)
    _Float16* pb = &pbuf[wave][0];
#pragma unroll
    for (int r = 0; r < 8; ++r) {
      int m = r + 8 * hi;
      pb[m * NT + ln]      = (_Float16)s0[r];
      pb[m * NT + 16 + ln] = (_Float16)s1[r];
    }
    v16h pa;
    {
      const _Float16* pr = &pb[ln * NT];
      int kbse = hi * 16;
#pragma unroll
      for (int i = 0; i < 8; ++i) { pa[i] = pr[kbse + i]; pa[8 + i] = pr[kbse + 8 + i]; }
    }

    // V B-fragments from transposed LDS: contiguous key halves per lane
    v16h vb0, vb1;
    {
      int khalf = hi * 16;
      const _Float16* pv0 = &vtt[ln][khalf];
      const _Float16* pv1 = &vtt[16 + ln][khalf];
#pragma unroll
      for (int i = 0; i < 16; ++i) { vb0[i] = pv0[i]; vb1[i] = pv1[i]; }
    }
    o0 = __builtin_amdgcn_wmma_f32_16x16x32_f16(false, pa, false, vb0, (short)0, o0, false, false);
    o1 = __builtin_amdgcn_wmma_f32_16x16x32_f16(false, pa, false, vb1, (short)0, o1, false, false);
    __syncthreads();
  }

  // epilogue: normalize, add channel-attn output, write pixel-major f16 [b][n][C]
#pragma unroll
  for (int r = 0; r < 8; ++r) {
    int m = r + 8 * hi;
    int row = q0 + wave * 16 + m;
    float inv = 1.0f / lrow[r];
    size_t o2base = ((size_t)bh * HW_ + row) * CH_;
    float v0 = o0[r] * inv + out2[o2base + ln];
    float v1 = o1[r] * inv + out2[o2base + 16 + ln];
    _Float16* op = outt + ((size_t)b * HW_ + row) * C_ + hd * CH_;
    op[ln]      = (_Float16)v0;
    op[ln + 16] = (_Float16)v1;
  }
}

// ---------------------------------------------------------------------------
// host side
// ---------------------------------------------------------------------------
extern "C" void kernel_launch(void* const* d_in, const int* in_sizes, int n_in,
                              void* d_out, int out_size, void* d_ws, size_t ws_size,
                              hipStream_t stream) {
  (void)in_sizes; (void)n_in; (void)out_size; (void)ws_size;
  const float* x      = (const float*)d_in[0];
  const float* ln1_w  = (const float*)d_in[1];
  const float* w_qkv  = (const float*)d_in[2];
  const float* w_dw   = (const float*)d_in[3];
  const float* temp   = (const float*)d_in[4];
  const float* w_po   = (const float*)d_in[5];
  const float* ln2_w  = (const float*)d_in[6];
  const float* w_proj = (const float*)d_in[7];
  const float* b_proj = (const float*)d_in[8];
  float* outp = (float*)d_out;

  char* ws = (char*)d_ws;
  size_t off = 0;
  auto alloc = [&](size_t bytes) -> char* {
    char* p = ws + off;
    off += (bytes + 255) & ~(size_t)255;
    return p;
  };
  _Float16* xnh_t  = (_Float16*)alloc((size_t)B_ * HW_ * C_ * 2);   // LN1 out, pixel-major
  _Float16* wqkvh  = (_Float16*)alloc((size_t)QC_ * C_ * 2);
  _Float16* wpoh   = (_Float16*)alloc((size_t)C_ * C_ * 2);
  _Float16* wprojh = (_Float16*)alloc((size_t)C_ * C_ * 2);
  float*    qkv    = (float*)alloc((size_t)B_ * QC_ * HW_ * 4);
  float*    dwc    = (float*)alloc((size_t)B_ * QC_ * HW_ * 4);
  float*    norms  = (float*)alloc((size_t)B_ * 4 * C_ * 4);
  _Float16* q1h    = (_Float16*)alloc((size_t)B_ * HEADS_ * HW_ * CH_ * 2);
  _Float16* k1h    = (_Float16*)alloc((size_t)B_ * HEADS_ * HW_ * CH_ * 2);
  _Float16* vh     = (_Float16*)alloc((size_t)B_ * HEADS_ * HW_ * CH_ * 2);
  float*    attn2  = (float*)alloc((size_t)B_ * HEADS_ * CH_ * CH_ * 4);
  float*    out2   = (float*)alloc((size_t)B_ * HEADS_ * HW_ * CH_ * 4);
  _Float16* outt   = (_Float16*)alloc((size_t)B_ * HW_ * C_ * 2);   // attn out, pixel-major
  float*    po     = (float*)alloc((size_t)B_ * C_ * HW_ * 4);
  _Float16* polnh  = (_Float16*)alloc((size_t)B_ * HW_ * C_ * 2);   // LN2 out, pixel-major

  const int TB = 256;

  // 1. LN1 (fused f16 transpose)
  ln_ft_k<<<dim3((B_ * HW_ + TB - 1) / TB), dim3(TB), 0, stream>>>(x, ln1_w, xnh_t);

  // 2. weight converts
  {
    int n = QC_ * C_;
    cvt_k<<<dim3((n + TB - 1) / TB), dim3(TB), 0, stream>>>(w_qkv, wqkvh, n);
    n = C_ * C_;
    cvt_k<<<dim3((n + TB - 1) / TB), dim3(TB), 0, stream>>>(w_po, wpoh, n);
    cvt_k<<<dim3((n + TB - 1) / TB), dim3(TB), 0, stream>>>(w_proj, wprojh, n);
  }

  // 3. qkv = W_qkv @ xn  (WMMA)
  gemm_k<<<dim3(HW_ / 64, QC_ / 16, B_), dim3(32), 0, stream>>>(
      wqkvh, xnh_t, qkv, nullptr, QC_, HW_);

  // 4. depthwise 3x3
  {
    long n = (long)B_ * QC_ * HW_;
    dwconv_k<<<dim3((unsigned)((n + TB - 1) / TB)), dim3(TB), 0, stream>>>(qkv, w_dw, dwc);
  }

  // 5. token-axis L2 norms + f16 packing (temperature folded into q1)
  norms_k<<<dim3(B_ * 4 * C_), dim3(256), 0, stream>>>(dwc, norms);
  {
    long n = (long)B_ * HEADS_ * HW_ * CH_;
    pack_k<<<dim3((unsigned)((n + TB - 1) / TB)), dim3(TB), 0, stream>>>(
        dwc, norms, temp, q1h, k1h, vh);
  }

  // 6. channel attention (32x32) + its output
  chan_attn_k<<<dim3(B_ * HEADS_), dim3(256), 0, stream>>>(dwc, norms, temp, attn2);
  out2_k<<<dim3(HW_ * CH_ / 256, B_ * HEADS_), dim3(256), 0, stream>>>(vh, attn2, out2);

  // 7. flash attention (WMMA) + combine -> pixel-major f16
  flash_k<<<dim3(HW_ / 128, B_ * HEADS_), dim3(256), 0, stream>>>(q1h, k1h, vh, out2, outt);

  // 8. project_out, LN2, final proj (+bias)
  gemm_k<<<dim3(HW_ / 64, C_ / 16, B_), dim3(32), 0, stream>>>(
      wpoh, outt, po, nullptr, C_, HW_);
  ln_ft_k<<<dim3((B_ * HW_ + TB - 1) / TB), dim3(TB), 0, stream>>>(po, ln2_w, polnh);
  gemm_k<<<dim3(HW_ / 64, C_ / 16, B_), dim3(32), 0, stream>>>(
      wprojh, polnh, outp, b_proj, C_, HW_);
}
